// single_head_Attention_15212774163203
// MI455X (gfx1250) — matrix operand order repaired and checked by reading, not compiled, not run on hardware
//
#include <hip/hip_runtime.h>

// Problem constants (from reference): B=8, S=2048, D=1024
#define Bn 8
#define Sn 2048
#define Dn 1024
#define BSn (Bn * Sn)
#define SCALE 0.03125f   // 1/sqrt(1024)

// LDS tile geometry: block tile 128x128, K-stage 64, padded row stride 72
#define KSTAGE 64
#define LDSTRIDE 72
#define BUFE (128 * LDSTRIDE)   // elements per LDS buffer (18 KB as bf16)

typedef __bf16 bf16;
typedef __attribute__((ext_vector_type(16))) __bf16 v16bf;
typedef __attribute__((ext_vector_type(8)))  float  v8f;

#define WMMA_BF16(a, b, c) \
  __builtin_amdgcn_wmma_f32_16x16x32_bf16(false, (a), false, (b), (short)0, (c), false, false)

// ---------------------------------------------------------------------------
// Async staging: copy a 128 x 64 bf16 tile (row-major, leading dim `ld`) into
// LDS (row stride LDSTRIDE) with per-lane global_load_async_to_lds_b128.
// Each thread issues 4 x 16B chunks; per wave that is 4 ASYNCcnt increments.
// ---------------------------------------------------------------------------
__device__ __forceinline__ void async_tile(const bf16* __restrict__ g, int ld,
                                           int k0, const bf16* lds_base, int tid) {
#pragma unroll
  for (int i = 0; i < 4; ++i) {
    const int c   = tid + (i << 8);       // 0..1023 chunk id
    const int row = c >> 3;               // 0..127
    const int cc  = (c & 7) << 3;         // 0,8,...,56 (bf16 elems)
    const unsigned long long src =
        (unsigned long long)(const void*)(g + (size_t)row * ld + (k0 + cc));
    const unsigned dst = (unsigned)(size_t)(const void*)(lds_base + row * LDSTRIDE + cc);
    asm volatile("global_load_async_to_lds_b128 %0, %1, off"
                 :: "v"(dst), "v"(src) : "memory");
  }
}

// ---------------------------------------------------------------------------
// Fragment loaders per CDNA5 ISA 7.12.2 (wave32), reading from LDS tiles.
// A (16x32): lane holds row M=lane&15; elems 0..7 -> K=(lane>>4)*8+0..7,
//            elems 8..15 -> K=16+(lane>>4)*8+0..7
// B (32x16) from Bt layout (row = output col): lane holds col N=lane&15;
//            elems 0..7 -> K=(lane>>4)*8+0..7, elems 8..15 -> K=16+(lane>>4)*8+0..7
// ---------------------------------------------------------------------------
__device__ __forceinline__ v16bf frag_a(const bf16* t, int rowbase, int kk, int lane) {
  const bf16* q = t + (rowbase + (lane & 15)) * LDSTRIDE + kk + ((lane >> 4) << 3);
  v16bf a;
#pragma unroll
  for (int i = 0; i < 8; ++i) { a[i] = q[i]; a[i + 8] = q[16 + i]; }
  return a;
}

__device__ __forceinline__ v16bf frag_b(const bf16* t, int rowbase, int kk, int lane) {
  const bf16* q = t + (rowbase + (lane & 15)) * LDSTRIDE + kk + ((lane >> 4) << 3);
  v16bf b;
#pragma unroll
  for (int i = 0; i < 8; ++i) { b[i] = q[i]; b[i + 8] = q[16 + i]; }
  return b;
}

// ---------------------------------------------------------------------------
// Shared GEMM core: C(128x128) += A(128xK) * Bt(128xK)^T with double-buffered
// async LDS staging. 8 waves; wave tile 32x64 (2x4 WMMA accumulators).
// ---------------------------------------------------------------------------
__device__ __forceinline__ void gemm_core(const bf16* __restrict__ Ag, int lda,
                                          const bf16* __restrict__ Bg, int ldb,
                                          int Kdim, bf16* As, bf16* Bs,
                                          v8f (&acc)[2][4]) {
  const int tid  = threadIdx.x;
  const int lane = tid & 31;
  const int wave = tid >> 5;
  const int wm = (wave & 3) * 32;   // wave row offset inside 128-tile
  const int wn = (wave >> 2) * 64;  // wave col offset inside 128-tile

  async_tile(Ag, lda, 0, As, tid);
  async_tile(Bg, ldb, 0, Bs, tid);

  for (int k0 = 0; k0 < Kdim; k0 += KSTAGE) {
    const int cur = (k0 >> 6) & 1;
    if (k0 + KSTAGE < Kdim) {
      async_tile(Ag, lda, k0 + KSTAGE, As + (cur ^ 1) * BUFE, tid);
      async_tile(Bg, ldb, k0 + KSTAGE, Bs + (cur ^ 1) * BUFE, tid);
      // 8 just-issued pending; wait until the current stage's 8 are done
      asm volatile("s_wait_asynccnt 0x8" ::: "memory");
    } else {
      asm volatile("s_wait_asynccnt 0x0" ::: "memory");
    }
    __syncthreads();  // current-stage LDS visible to all waves

    const bf16* At = As + cur * BUFE;
    const bf16* Bt = Bs + cur * BUFE;
#pragma unroll
    for (int kk = 0; kk < KSTAGE; kk += 32) {
      v16bf a[2], b[4];
#pragma unroll
      for (int mf = 0; mf < 2; ++mf) a[mf] = frag_a(At, wm + mf * 16, kk, lane);
#pragma unroll
      for (int nf = 0; nf < 4; ++nf) b[nf] = frag_b(Bt, wn + nf * 16, kk, lane);
#pragma unroll
      for (int mf = 0; mf < 2; ++mf)
#pragma unroll
        for (int nf = 0; nf < 4; ++nf)
          acc[mf][nf] = WMMA_BF16(a[mf], b[nf], acc[mf][nf]);
    }
    __syncthreads();  // protect buffer before next refill
  }
}

// ---------------------------------------------------------------------------
// Kernel 0: elementwise convert X f32 -> bf16
// ---------------------------------------------------------------------------
__global__ void k_convX(const float* __restrict__ X, bf16* __restrict__ Xb) {
  const size_t i = ((size_t)blockIdx.x * 256 + threadIdx.x) * 4;
  const float4 v = *(const float4*)(X + i);
  Xb[i + 0] = (bf16)v.x;
  Xb[i + 1] = (bf16)v.y;
  Xb[i + 2] = (bf16)v.z;
  Xb[i + 3] = (bf16)v.w;
}

// ---------------------------------------------------------------------------
// Kernel 1: transpose+convert one DxD f32 weight -> bf16 Wt[n][k] = W[k][n]
// ---------------------------------------------------------------------------
__global__ void k_wt_transpose(const float* __restrict__ W, bf16* __restrict__ Wt) {
  __shared__ float tile[32][33];
  int x = blockIdx.x * 32 + threadIdx.x;  // n
  int y = blockIdx.y * 32 + threadIdx.y;  // k
#pragma unroll
  for (int j = 0; j < 32; j += 8)
    tile[threadIdx.y + j][threadIdx.x] = W[(size_t)(y + j) * Dn + x];
  __syncthreads();
  int x2 = blockIdx.y * 32 + threadIdx.x;  // k
  int y2 = blockIdx.x * 32 + threadIdx.y;  // n
#pragma unroll
  for (int j = 0; j < 32; j += 8)
    Wt[(size_t)(y2 + j) * Dn + x2] = (bf16)tile[threadIdx.x][threadIdx.y + j];
}

// ---------------------------------------------------------------------------
// Kernel 2: fused QKV projection GEMM. blockIdx.z selects Q / K / V.
// Q,K stored bf16 row-major [BS][D]; V stored transposed Vt[b][d][s].
// ---------------------------------------------------------------------------
__global__ void k_proj(const bf16* __restrict__ Xb, const bf16* __restrict__ Wt_all,
                       bf16* __restrict__ Qb, bf16* __restrict__ Kb, bf16* __restrict__ Vt) {
  __shared__ bf16 As[2 * BUFE];
  __shared__ bf16 Bs[2 * BUFE];
  const int mode = blockIdx.z;
  const int gm = blockIdx.x * 128;  // row base in [0, BS)
  const int gn = blockIdx.y * 128;  // col base in [0, D)
  const int lane = threadIdx.x & 31;
  const int wave = threadIdx.x >> 5;
  const int wm = gm + (wave & 3) * 32;
  const int wn = gn + (wave >> 2) * 64;

  v8f acc[2][4] = {};
  gemm_core(Xb + (size_t)gm * Dn, Dn,
            Wt_all + (size_t)mode * Dn * Dn + (size_t)gn * Dn, Dn,
            Dn, As, Bs, acc);

  if (mode == 2) {
    // V transposed store: Vt[b][n][m_local]; consecutive r -> contiguous m.
    const int bidx = wm / Sn;
    const int mloc = wm - bidx * Sn;
#pragma unroll
    for (int mf = 0; mf < 2; ++mf)
#pragma unroll
      for (int nf = 0; nf < 4; ++nf) {
        const int n = wn + nf * 16 + (lane & 15);
        const int m = mloc + mf * 16 + ((lane >> 4) << 3);
        bf16* dst = Vt + ((size_t)bidx * Dn + n) * Sn + m;
#pragma unroll
        for (int r = 0; r < 8; ++r) dst[r] = (bf16)acc[mf][nf][r];
      }
  } else {
    bf16* Y = (mode == 0) ? Qb : Kb;
#pragma unroll
    for (int mf = 0; mf < 2; ++mf)
#pragma unroll
      for (int nf = 0; nf < 4; ++nf) {
        const int n = wn + nf * 16 + (lane & 15);
        const int mb = wm + mf * 16 + ((lane >> 4) << 3);
#pragma unroll
        for (int r = 0; r < 8; ++r)
          Y[(size_t)(mb + r) * Dn + n] = (bf16)acc[mf][nf][r];
      }
  }
}

// ---------------------------------------------------------------------------
// Kernel 3: scores GEMM  Sb[b][q][kk] = (Q·Kt)/32  (bf16 out)
// ---------------------------------------------------------------------------
__global__ void k_scores(const bf16* __restrict__ Qb, const bf16* __restrict__ Kb,
                         bf16* __restrict__ Sb) {
  __shared__ bf16 As[2 * BUFE];
  __shared__ bf16 Bs[2 * BUFE];
  const int bidx = blockIdx.z;
  const bf16* Q = Qb + (size_t)bidx * Sn * Dn;
  const bf16* K = Kb + (size_t)bidx * Sn * Dn;
  bf16* S = Sb + (size_t)bidx * Sn * Sn;
  const int gq = blockIdx.x * 128, gk = blockIdx.y * 128;
  const int lane = threadIdx.x & 31;
  const int wave = threadIdx.x >> 5;
  const int wq = gq + (wave & 3) * 32;
  const int wk = gk + (wave >> 2) * 64;

  v8f acc[2][4] = {};
  gemm_core(Q + (size_t)gq * Dn, Dn, K + (size_t)gk * Dn, Dn, Dn, As, Bs, acc);

#pragma unroll
  for (int mf = 0; mf < 2; ++mf)
#pragma unroll
    for (int nf = 0; nf < 4; ++nf) {
      const int n = wk + nf * 16 + (lane & 15);
      const int mb = wq + mf * 16 + ((lane >> 4) << 3);
#pragma unroll
      for (int r = 0; r < 8; ++r)
        S[(size_t)(mb + r) * Sn + n] = (bf16)(acc[mf][nf][r] * SCALE);
    }
}

// ---------------------------------------------------------------------------
// Kernel 4: softmax, one wave32 per row of 2048, in place on Sb (bf16).
// ---------------------------------------------------------------------------
__global__ void k_softmax(bf16* __restrict__ Sb) {
  const int row = blockIdx.x * 8 + (threadIdx.x >> 5);
  const int lane = threadIdx.x & 31;
  bf16* p = Sb + (size_t)row * Sn;

  float v[64];
  float mx = -3.0e38f;
#pragma unroll
  for (int i = 0; i < 64; ++i) {
    v[i] = (float)p[i * 32 + lane];
    mx = fmaxf(mx, v[i]);
  }
#pragma unroll
  for (int off = 16; off > 0; off >>= 1)
    mx = fmaxf(mx, __shfl_xor(mx, off, 32));
  float sum = 0.0f;
#pragma unroll
  for (int i = 0; i < 64; ++i) { v[i] = __expf(v[i] - mx); sum += v[i]; }
#pragma unroll
  for (int off = 16; off > 0; off >>= 1)
    sum += __shfl_xor(sum, off, 32);
  const float inv = 1.0f / sum;
#pragma unroll
  for (int i = 0; i < 64; ++i) p[i * 32 + lane] = (bf16)(v[i] * inv);
}

// ---------------------------------------------------------------------------
// Kernel 5: out GEMM  out[b][q][d] = sum_k P[b][q][k] * Vt[b][d][k]  (f32 out)
// ---------------------------------------------------------------------------
__global__ void k_out(const bf16* __restrict__ Sb, const bf16* __restrict__ Vt,
                      float* __restrict__ out) {
  __shared__ bf16 As[2 * BUFE];
  __shared__ bf16 Bs[2 * BUFE];
  const int bidx = blockIdx.z;
  const bf16* P = Sb + (size_t)bidx * Sn * Sn;
  const bf16* V = Vt + (size_t)bidx * Dn * Sn;
  const int gq = blockIdx.x * 128, gd = blockIdx.y * 128;
  const int lane = threadIdx.x & 31;
  const int wave = threadIdx.x >> 5;
  const int wq = gq + (wave & 3) * 32;
  const int wd = gd + (wave >> 2) * 64;

  v8f acc[2][4] = {};
  gemm_core(P + (size_t)gq * Sn, Sn, V + (size_t)gd * Sn, Sn, Sn, As, Bs, acc);

#pragma unroll
  for (int mf = 0; mf < 2; ++mf)
#pragma unroll
    for (int nf = 0; nf < 4; ++nf) {
      const int n = wd + nf * 16 + (lane & 15);
      const int mb = wq + mf * 16 + ((lane >> 4) << 3);
#pragma unroll
      for (int r = 0; r < 8; ++r)
        out[((size_t)bidx * Sn + mb + r) * Dn + n] = acc[mf][nf][r];
    }
}

// ---------------------------------------------------------------------------
// Host launcher. Workspace layout (bf16 elements):
//   Xb (BS*D) | Wt (3*D*D) | Qb (BS*D) | Kb (BS*D) | Vt (B*D*S) | Sb (B*S*S)
// Total ~208 MB.
// ---------------------------------------------------------------------------
extern "C" void kernel_launch(void* const* d_in, const int* in_sizes, int n_in,
                              void* d_out, int out_size, void* d_ws, size_t ws_size,
                              hipStream_t stream) {
  const float* X  = (const float*)d_in[0];
  const float* Wq = (const float*)d_in[1];
  const float* Wk = (const float*)d_in[2];
  const float* Wv = (const float*)d_in[3];
  float* out = (float*)d_out;

  bf16* Xb = (bf16*)d_ws;
  bf16* Wt = Xb + (size_t)BSn * Dn;
  bf16* Qb = Wt + (size_t)3 * Dn * Dn;
  bf16* Kb = Qb + (size_t)BSn * Dn;
  bf16* Vt = Kb + (size_t)BSn * Dn;
  bf16* Sb = Vt + (size_t)Bn * Dn * Sn;

  // 0) input convert f32 -> bf16
  k_convX<<<(size_t)BSn * Dn / (256 * 4), 256, 0, stream>>>(X, Xb);
  // 1) weight transpose+convert
  {
    dim3 tb(32, 8), tg(Dn / 32, Dn / 32);
    k_wt_transpose<<<tg, tb, 0, stream>>>(Wq, Wt + (size_t)0 * Dn * Dn);
    k_wt_transpose<<<tg, tb, 0, stream>>>(Wk, Wt + (size_t)1 * Dn * Dn);
    k_wt_transpose<<<tg, tb, 0, stream>>>(Wv, Wt + (size_t)2 * Dn * Dn);
  }
  // 2) fused QKV projections
  {
    dim3 g(BSn / 128, Dn / 128, 3);
    k_proj<<<g, 256, 0, stream>>>(Xb, Wt, Qb, Kb, Vt);
  }
  // 3) scores
  {
    dim3 g(Sn / 128, Sn / 128, Bn);
    k_scores<<<g, 256, 0, stream>>>(Qb, Kb, Sb);
  }
  // 4) softmax (wave per row)
  k_softmax<<<(BSn) / 8, 256, 0, stream>>>(Sb);
  // 5) output GEMM
  {
    dim3 g(Sn / 128, Dn / 128, Bn);
    k_out<<<g, 256, 0, stream>>>(Sb, Vt, out);
  }
}
